// MultiheadAttention_49340584296500
// MI455X (gfx1250) — compile-verified, matrix operand order checked by reading, not run on hardware
//
#include <hip/hip_runtime.h>

typedef __bf16 bf16;
typedef __attribute__((ext_vector_type(16))) __bf16 v16bf;
typedef __attribute__((ext_vector_type(8)))  __bf16 v8bf;
typedef __attribute__((ext_vector_type(8)))  float  v8f;

#define EMBED 1024
#define SEQ   2048
#define HEADS_N 16
#define HDIM  64
#define ROWS  8192        // B*S

// ---------------------------------------------------------------------------
// WMMA helpers
// ---------------------------------------------------------------------------
__device__ __forceinline__ v8f wmma_bf16(v16bf a, v16bf b, v8f c) {
  return __builtin_amdgcn_wmma_f32_16x16x32_bf16(false, a, false, b, (short)0, c,
                                                 false, false);
}

// Load a 16x32 bf16 A/B fragment from an LDS tile stored [row][K] with row
// stride `ld` (elements, multiple of 8 -> 16B aligned chunks).
// Lane L (0-15): row = row_base+L, K chunks {k0..k0+7} and {k0+16..k0+23}.
// Lane L+16:     row = row_base+L, K chunks {k0+8..} and {k0+24..}.
__device__ __forceinline__ v16bf load_frag(const bf16* p, int row, int k0,
                                           int ld, int lane) {
  const int half = lane >> 4;
  const bf16* b0 = p + row * ld + k0 + half * 8;
  union { v16bf v; v8bf h[2]; } u;
  u.h[0] = *(const v8bf*)b0;
  u.h[1] = *(const v8bf*)(b0 + 16);
  return u.v;
}

__device__ __forceinline__ v8f zero8() {
  v8f z = {0.f, 0.f, 0.f, 0.f, 0.f, 0.f, 0.f, 0.f};
  return z;
}

// ---------------------------------------------------------------------------
// GEMM: C[M,N] = (A[M,K] @ W[N,K]^T + bias) * scale
//   ABF16=false: A is fp32 (converted to bf16 on LDS fill)
//   HEADS=true : store bf16 to out in [B,H,S,D] layout; else fp32 row-major
// Block 256 threads (8 waves), tile 128x128, K-step 64, register-pipelined
// global loads (stage tile k+1 in VGPRs while computing on LDS tile k).
// ---------------------------------------------------------------------------
#define BM 128
#define BN 128
#define BK 64
#define LDT 72   // LDS row stride (bf16 elems): mult of 8, staggers banks

template <bool HEADS, bool ABF16>
__global__ __launch_bounds__(256) void wmma_gemm(
    const float* __restrict__ A32, const bf16* __restrict__ A16,
    const float* __restrict__ W, const float* __restrict__ bias,
    float* __restrict__ outF, bf16* __restrict__ outH,
    int M, int N, int Kd, float scale) {
  __shared__ alignas(16) bf16 As[BM * LDT];
  __shared__ alignas(16) bf16 Bs[BN * LDT];

  const int tid  = threadIdx.x;
  const int lane = tid & 31;
  const int wave = tid >> 5;
  const int l16  = lane & 15;
  const int half = lane >> 4;
  const int wm   = wave & 1;   // 2 waves over M (64 rows each)
  const int wn   = wave >> 1;  // 4 waves over N (32 cols each)
  const int m0   = blockIdx.x * BM;
  const int n0   = blockIdx.y * BN;

  // staging coordinates: each thread owns half a row of the 128x64 tile
  const int lr = tid >> 1;          // 0..127
  const int lc = (tid & 1) * 32;    // 0 or 32

  float4 aS[8];      // fp32 A staging (32 floats)
  v8bf   aS16[4];    // bf16 A staging (32 bf16)
  float4 wS[8];      // fp32 W staging

  auto loadRegs = [&](int k0) {
    if constexpr (!ABF16) {
#pragma unroll
      for (int j = 0; j < 8; ++j)
        aS[j] = *(const float4*)(A32 + (size_t)(m0 + lr) * Kd + k0 + lc + j * 4);
    } else {
#pragma unroll
      for (int j = 0; j < 4; ++j)
        aS16[j] = *(const v8bf*)(A16 + (size_t)(m0 + lr) * Kd + k0 + lc + j * 8);
    }
#pragma unroll
    for (int j = 0; j < 8; ++j)
      wS[j] = *(const float4*)(W + (size_t)(n0 + lr) * Kd + k0 + lc + j * 4);
  };

  auto storeRegs = [&]() {
    if constexpr (!ABF16) {
#pragma unroll
      for (int j = 0; j < 8; ++j) {
        bf16* d = As + lr * LDT + lc + j * 4;
        d[0] = (bf16)aS[j].x; d[1] = (bf16)aS[j].y;
        d[2] = (bf16)aS[j].z; d[3] = (bf16)aS[j].w;
      }
    } else {
#pragma unroll
      for (int j = 0; j < 4; ++j)
        *(v8bf*)(As + lr * LDT + lc + j * 8) = aS16[j];
    }
#pragma unroll
    for (int j = 0; j < 8; ++j) {
      bf16* d = Bs + lr * LDT + lc + j * 4;
      d[0] = (bf16)wS[j].x; d[1] = (bf16)wS[j].y;
      d[2] = (bf16)wS[j].z; d[3] = (bf16)wS[j].w;
    }
  };

  v8f acc[4][2];
#pragma unroll
  for (int i = 0; i < 4; ++i) { acc[i][0] = zero8(); acc[i][1] = zero8(); }

  loadRegs(0);
  for (int k0 = 0; k0 < Kd; k0 += BK) {
    storeRegs();
    __syncthreads();
    if (k0 + BK < Kd) {
      loadRegs(k0 + BK);                       // hide HBM latency behind WMMAs
      if (k0 + 2 * BK < Kd) {                  // pull tile k+2 into L2
        __builtin_prefetch(W + (size_t)(n0 + lr) * Kd + k0 + 2 * BK + lc, 0, 1);
        if constexpr (!ABF16)
          __builtin_prefetch(A32 + (size_t)(m0 + lr) * Kd + k0 + 2 * BK + lc, 0, 1);
        else
          __builtin_prefetch(A16 + (size_t)(m0 + lr) * Kd + k0 + 2 * BK + lc, 0, 1);
      }
    }
#pragma unroll
    for (int kf = 0; kf < BK; kf += 32) {
      v16bf b0 = load_frag(Bs, wn * 32 + l16, kf, LDT, lane);
      v16bf b1 = load_frag(Bs, wn * 32 + 16 + l16, kf, LDT, lane);
#pragma unroll
      for (int mi = 0; mi < 4; ++mi) {
        v16bf a = load_frag(As, wm * 64 + mi * 16 + l16, kf, LDT, lane);
        acc[mi][0] = wmma_bf16(a, b0, acc[mi][0]);
        acc[mi][1] = wmma_bf16(a, b1, acc[mi][1]);
      }
    }
    __syncthreads();
  }

  // ---- epilogue: bias + scale + store ----
#pragma unroll
  for (int mi = 0; mi < 4; ++mi) {
#pragma unroll
    for (int ni = 0; ni < 2; ++ni) {
      int n = n0 + wn * 32 + ni * 16 + l16;
      float bv = bias[n];
#pragma unroll
      for (int r = 0; r < 8; ++r) {
        int m = m0 + wm * 64 + mi * 16 + r + 8 * half;
        float val = (acc[mi][ni][r] + bv) * scale;
        if (HEADS) {
          int b = m >> 11, s = m & (SEQ - 1);     // S = 2048
          int h = n >> 6, d = n & (HDIM - 1);     // D = 64
          outH[(((size_t)(b * HEADS_N + h) * SEQ + s) << 6) + d] = (bf16)val;
        } else {
          outF[(size_t)m * N + n] = val;
        }
      }
    }
  }
}

// ---------------------------------------------------------------------------
// Flash attention over bf16 Q/K/V in [B,H,S,D] layout, ctx -> [B,S,H*D] bf16.
// Q is pre-scaled by 1/sqrt(D) in the projection epilogue.
// Block: 256 threads / 8 waves; 128 queries per block; 64-key tiles,
// register-pipelined K/V global loads.
// ---------------------------------------------------------------------------
__global__ __launch_bounds__(256) void flash_attn(
    const bf16* __restrict__ Q, const bf16* __restrict__ Kg,
    const bf16* __restrict__ V, bf16* __restrict__ ctx) {
  __shared__ alignas(16) bf16 Qs[128 * 72];
  __shared__ alignas(16) bf16 Ks[64 * 72];
  __shared__ alignas(16) bf16 Vt[64 * 72];       // Vt[d][key]
  __shared__ alignas(16) bf16 Ps[8][16 * 72];    // per-wave P scratch

  const int tid  = threadIdx.x;
  const int lane = tid & 31;
  const int wave = tid >> 5;
  const int l16  = lane & 15;
  const int half = lane >> 4;
  const int bh   = blockIdx.y;          // b*16 + h
  const int q0   = blockIdx.x * 128;

  const bf16* Qb = Q  + (size_t)bh * SEQ * HDIM;
  const bf16* Kb = Kg + (size_t)bh * SEQ * HDIM;
  const bf16* Vb = V  + (size_t)bh * SEQ * HDIM;

  // load Q tile [128][64]
#pragma unroll
  for (int it = 0; it < 4; ++it) {
    int r = (tid >> 3) + it * 32;
    int c = (tid & 7) * 8;
    *(v8bf*)(Qs + r * 72 + c) = *(const v8bf*)(Qb + (size_t)(q0 + r) * HDIM + c);
  }

  v8bf kS[2], vS[2];
  auto loadKV = [&](int kt) {
#pragma unroll
    for (int it = 0; it < 2; ++it) {
      int r = (tid >> 3) + it * 32;
      int c = (tid & 7) * 8;
      kS[it] = *(const v8bf*)(Kb + (size_t)(kt + r) * HDIM + c);
      vS[it] = *(const v8bf*)(Vb + (size_t)(kt + r) * HDIM + c);
    }
  };
  auto storeKV = [&]() {
#pragma unroll
    for (int it = 0; it < 2; ++it) {
      int r = (tid >> 3) + it * 32;
      int c = (tid & 7) * 8;
      *(v8bf*)(Ks + r * 72 + c) = kS[it];
#pragma unroll
      for (int j = 0; j < 8; ++j) Vt[(c + j) * 72 + r] = vS[it][j];
    }
  };

  v8f o[4];
#pragma unroll
  for (int n = 0; n < 4; ++n) o[n] = zero8();
  float mrow[8], lrow[8];
#pragma unroll
  for (int r = 0; r < 8; ++r) { mrow[r] = -1e30f; lrow[r] = 0.f; }

  loadKV(0);
  for (int kt = 0; kt < SEQ; kt += 64) {
    __syncthreads();              // prior compute done (Qs loaded on iter 0)
    storeKV();
    __syncthreads();
    if (kt + 64 < SEQ) loadKV(kt + 64);   // stage next tile behind the math

    // ---- scores S = Q @ K^T (16 rows x 64 keys per wave) ----
    v8f sc[4];
#pragma unroll
    for (int n = 0; n < 4; ++n) sc[n] = zero8();
#pragma unroll
    for (int kf = 0; kf < 64; kf += 32) {
      v16bf a = load_frag(Qs, wave * 16 + l16, kf, 72, lane);
#pragma unroll
      for (int n = 0; n < 4; ++n) {
        v16bf b = load_frag(Ks, n * 16 + l16, kf, 72, lane);
        sc[n] = wmma_bf16(a, b, sc[n]);
      }
    }

    // ---- online softmax per row (row = r + 8*half within wave tile) ----
    float fac[8];
#pragma unroll
    for (int r = 0; r < 8; ++r) {
      float mloc = fmaxf(fmaxf(sc[0][r], sc[1][r]), fmaxf(sc[2][r], sc[3][r]));
#pragma unroll
      for (int off = 1; off < 16; off <<= 1)
        mloc = fmaxf(mloc, __shfl_xor(mloc, off, 32));
      float mnew = fmaxf(mrow[r], mloc);
      fac[r] = __expf(mrow[r] - mnew);
      mrow[r] = mnew;
      float ps = 0.f;
#pragma unroll
      for (int n = 0; n < 4; ++n) {
        float p = __expf(sc[n][r] - mnew);
        sc[n][r] = p;
        ps += p;
      }
#pragma unroll
      for (int off = 1; off < 16; off <<= 1) ps += __shfl_xor(ps, off, 32);
      lrow[r] = lrow[r] * fac[r] + ps;
#pragma unroll
      for (int n = 0; n < 4; ++n) o[n][r] *= fac[r];
      // stage P into per-wave LDS scratch (same-wave DS ops are in order)
#pragma unroll
      for (int n = 0; n < 4; ++n)
        Ps[wave][(r + 8 * half) * 72 + n * 16 + l16] = (bf16)sc[n][r];
    }

    // ---- O += P @ V ----
#pragma unroll
    for (int kf = 0; kf < 64; kf += 32) {
      v16bf a = load_frag(&Ps[wave][0], l16, kf, 72, lane);
#pragma unroll
      for (int n = 0; n < 4; ++n) {
        v16bf b = load_frag(Vt, n * 16 + l16, kf, 72, lane);
        o[n] = wmma_bf16(a, b, o[n]);
      }
    }
  }

  // ---- write ctx in [B,S,H*D] layout (bf16) ----
  const int b = bh >> 4, h = bh & 15;
#pragma unroll
  for (int r = 0; r < 8; ++r) {
    float inv = 1.f / lrow[r];
    int srow = q0 + wave * 16 + r + 8 * half;
#pragma unroll
    for (int n = 0; n < 4; ++n) {
      int d = n * 16 + l16;
      ctx[(size_t)(b * SEQ + srow) * EMBED + h * HDIM + d] = (bf16)(o[n][r] * inv);
    }
  }
}

// ---------------------------------------------------------------------------
// Host launcher
// ---------------------------------------------------------------------------
extern "C" void kernel_launch(void* const* d_in, const int* in_sizes, int n_in,
                              void* d_out, int out_size, void* d_ws,
                              size_t ws_size, hipStream_t stream) {
  const float* x  = (const float*)d_in[0];
  const float* Wq = (const float*)d_in[1];
  const float* bq = (const float*)d_in[2];
  const float* Wk = (const float*)d_in[3];
  const float* bk = (const float*)d_in[4];
  const float* Wv = (const float*)d_in[5];
  const float* bv = (const float*)d_in[6];
  const float* Wo = (const float*)d_in[7];
  const float* bo = (const float*)d_in[8];
  float* out = (float*)d_out;

  const size_t SZ = (size_t)ROWS * EMBED;  // 8M elems, 16 MB bf16 each
  bf16* ws  = (bf16*)d_ws;
  bf16* q   = ws;
  bf16* k   = ws + SZ;
  bf16* v   = ws + 2 * SZ;
  bf16* ctx = ws + 3 * SZ;

  const float qscale = 0.125f;  // 1/sqrt(HDIM) folded into Q projection

  dim3 block(256);
  dim3 gGemm(ROWS / BM, EMBED / BN);   // 64 x 8
  dim3 gAttn(SEQ / 128, 4 * HEADS_N);  // 16 x 64

  wmma_gemm<true, false><<<gGemm, block, 0, stream>>>(
      x, nullptr, Wq, bq, nullptr, q, ROWS, EMBED, EMBED, qscale);
  wmma_gemm<true, false><<<gGemm, block, 0, stream>>>(
      x, nullptr, Wk, bk, nullptr, k, ROWS, EMBED, EMBED, 1.0f);
  wmma_gemm<true, false><<<gGemm, block, 0, stream>>>(
      x, nullptr, Wv, bv, nullptr, v, ROWS, EMBED, EMBED, 1.0f);
  flash_attn<<<gAttn, block, 0, stream>>>(q, k, v, ctx);
  wmma_gemm<false, true><<<gGemm, block, 0, stream>>>(
      nullptr, ctx, Wo, bo, out, nullptr, ROWS, EMBED, EMBED, 1.0f);
}
